// StructuralGraphTower_52192442581362
// MI455X (gfx1250) — matrix operand order repaired and checked by reading, not compiled, not run on hardware
//
#include <hip/hip_runtime.h>
#include <hip/hip_bf16.h>

typedef __attribute__((ext_vector_type(16))) __bf16 v16bf;
typedef __attribute__((ext_vector_type(8)))  float  v8f;

#define Nn   16384
#define En   262144
#define HIDn 1024
#define Dn   768
#define Rn   6
#define Bb   4
#define Ln   2

union FragU { uint4 q[2]; v16bf v; };

// LDS layout: rows padded to 80B (64B data + 16B pad) to spread ds banks.
#define ROWB   80
#define TILE_B (128 * ROWB)          // one A or B tile = 10240 B
// smem: [buf0.A][buf0.B][buf1.A][buf1.B] = 40960 B

__device__ __forceinline__ void async_b128(unsigned lds_addr, const void* gptr)
{
    asm volatile("global_load_async_to_lds_b128 %0, %1, off"
                 :: "v"(lds_addr), "v"(gptr) : "memory");
}

// ---------------------------------------------------------------------------
// bf16 WMMA GEMM: C[M x Nc] = A[M x K] @ Bt[Nc x K]^T (+bias)(+relu)
// 256 threads = 8 waves. Block tile 128x128, wave tile 32x64 (8 accumulators).
// A/B tiles double-buffered in LDS via global_load_async_to_lds_b128.
// Batched over blockIdx.z (strides on Bt and C).
// ---------------------------------------------------------------------------
__global__ __launch_bounds__(256)
void wmma_gemm_bf16(const __bf16* __restrict__ A, const __bf16* __restrict__ Bt,
                    float* __restrict__ C, const float* __restrict__ bias,
                    int M, int Nc, int K, long strideB, long strideC, int relu)
{
    __shared__ __align__(16) char smem[4 * TILE_B];

    const int t     = threadIdx.x;
    const int lane  = t & 31;
    const int wave  = t >> 5;
    const int l15   = lane & 15;
    const int hsel  = lane >> 4;
    const int waveM = wave >> 1;       // 0..3 -> 32-row strip
    const int waveN = wave & 1;        // 0..1 -> 64-col strip

    const int rowBase = blockIdx.y * 128;
    const int colBase = blockIdx.x * 128;
    const __bf16* Ab  = A  + (long)rowBase * K;
    const __bf16* Bb_ = Bt + (long)blockIdx.z * strideB + (long)colBase * K;
    float*        Cz  = C  + (long)blockIdx.z * strideC;

    const unsigned smemBase = (unsigned)(size_t)(void*)&smem[0];

    const v8f vzero = {0.f,0.f,0.f,0.f,0.f,0.f,0.f,0.f};
    v8f acc[8];
#pragma unroll
    for (int j = 0; j < 8; ++j) acc[j] = vzero;

    const int nStages = K >> 5;        // K / 32  (>= 2 for all our GEMMs)

    // Per-thread async-copy chunk coordinates (2 A + 2 B chunks of 16B/stage).
    // chunk c in [0,512): row = c>>2, 16B sub-chunk = c&3. Pointers advance
    // by 32 elements (64B) each stage.
    const int row0 = t >> 2,           kc0 = t & 3;
    const int row1 = (t + 256) >> 2,   kc1 = (t + 256) & 3;
    const __bf16* gA0 = Ab  + (long)row0 * K + kc0 * 8;
    const __bf16* gA1 = Ab  + (long)row1 * K + kc1 * 8;
    const __bf16* gB0 = Bb_ + (long)row0 * K + kc0 * 8;
    const __bf16* gB1 = Bb_ + (long)row1 * K + kc1 * 8;
    const unsigned lo0 = row0 * ROWB + kc0 * 16;
    const unsigned lo1 = row1 * ROWB + kc1 * 16;

    auto issue = [&](int buf) {
        const unsigned ldsA = smemBase + buf * 2 * TILE_B;
        const unsigned ldsB = ldsA + TILE_B;
        async_b128(ldsA + lo0, gA0);
        async_b128(ldsB + lo0, gB0);
        async_b128(ldsA + lo1, gA1);
        async_b128(ldsB + lo1, gB1);
        gA0 += 32; gA1 += 32; gB0 += 32; gB1 += 32;
    };

    auto compute = [&](int buf) {
        const int offA = buf * 2 * TILE_B;
        const int offB = offA + TILE_B;
        // A fragments for both 16-row subtiles (held in registers).
        FragU a0, a1;
        const int r0 = offA + (waveM * 32 + l15) * ROWB + hsel * 16;
        a0.q[0] = *(const uint4*)&smem[r0];
        a0.q[1] = *(const uint4*)&smem[r0 + 32];
        const int r1 = r0 + 16 * ROWB;
        a1.q[0] = *(const uint4*)&smem[r1];
        a1.q[1] = *(const uint4*)&smem[r1 + 32];
#pragma unroll
        for (int j = 0; j < 4; ++j) {
            FragU b;
            const int rb = offB + (waveN * 64 + j * 16 + l15) * ROWB + hsel * 16;
            b.q[0] = *(const uint4*)&smem[rb];
            b.q[1] = *(const uint4*)&smem[rb + 32];
            acc[j]     = __builtin_amdgcn_wmma_f32_16x16x32_bf16(
                             false, a0.v, false, b.v, (short)0, acc[j],     false, false);
            acc[4 + j] = __builtin_amdgcn_wmma_f32_16x16x32_bf16(
                             false, a1.v, false, b.v, (short)0, acc[4 + j], false, false);
        }
    };

    issue(0);
    // Main loop: straight-line issue -> wait(4) -> barrier -> 8x wmma -> barrier
    for (int s = 0; s < nStages - 1; ++s) {
        issue((s + 1) & 1);
        asm volatile("s_wait_asynccnt 0x4" ::: "memory");
        __syncthreads();
        compute(s & 1);
        __syncthreads();
    }
    // Peeled final stage
    asm volatile("s_wait_asynccnt 0x0" ::: "memory");
    __syncthreads();
    compute((nStages - 1) & 1);

#pragma unroll
    for (int m = 0; m < 2; ++m) {
#pragma unroll
        for (int j = 0; j < 4; ++j) {
            const int col = colBase + waveN * 64 + j * 16 + l15;
            const float bv = bias ? bias[col] : 0.f;
#pragma unroll
            for (int v = 0; v < 8; ++v) {
                const int row = rowBase + waveM * 32 + m * 16 + v + hsel * 8;
                float val = acc[m * 4 + j][v] + bv;
                if (relu) val = fmaxf(val, 0.f);
                Cz[(long)row * Nc + col] = val;
            }
        }
    }
}

// ---------------------------------------------------------------------------
// Elementwise / conversion kernels
// ---------------------------------------------------------------------------
__global__ void f32_to_bf16_k(const float* __restrict__ in, __bf16* __restrict__ out, long n)
{
    long i = (long)blockIdx.x * 256 + threadIdx.x;
    if (i < n) out[i] = (__bf16)in[i];
}

// out[c*rows + r] = (bf16) in[r*cols + c]   (transpose + convert)
__global__ void transpose_bf16_k(const float* __restrict__ in, __bf16* __restrict__ out,
                                 int rows, int cols)
{
    long i = (long)blockIdx.x * 256 + threadIdx.x;
    long n = (long)rows * cols;
    if (i < n) {
        int r = (int)(i / cols), c = (int)(i % cols);
        out[(long)c * rows + r] = (__bf16)in[i];
    }
}

__global__ void zero_f32_k(float* __restrict__ p, long n)
{
    long i = (long)blockIdx.x * 256 + threadIdx.x;
    if (i < n) p[i] = 0.f;
}

// Wt[r][o*D + i] = sum_b comp[r,b] * basis[b,i,o]   (pre-transposed for GEMM)
__global__ void build_rel_weights_k(const float* __restrict__ basis,
                                    const float* __restrict__ comp,
                                    __bf16* __restrict__ wt)
{
    long idx = (long)blockIdx.x * 256 + threadIdx.x;
    if (idx >= (long)Rn * Dn * Dn) return;
    int i = (int)(idx % Dn);
    int o = (int)((idx / Dn) % Dn);
    int r = (int)(idx / ((long)Dn * Dn));
    float s = 0.f;
#pragma unroll
    for (int b = 0; b < Bb; ++b)
        s += comp[r * Bb + b] * basis[((long)b * Dn + i) * Dn + o];
    wt[idx] = (__bf16)s;
}

__global__ void count_edges_k(const int* __restrict__ ei, const int* __restrict__ et,
                              float* __restrict__ counts)
{
    int e = blockIdx.x * 256 + threadIdx.x;
    if (e < En) atomicAdd(&counts[(long)et[e] * Nn + ei[En + e]], 1.0f);
}

__global__ void scatter_msgs_k(const int* __restrict__ ei, const int* __restrict__ et,
                               const float* __restrict__ h_all,
                               const float* __restrict__ counts,
                               float* __restrict__ agg)
{
    long idx = (long)blockIdx.x * 256 + threadIdx.x;
    if (idx >= (long)En * Dn) return;
    int e = (int)(idx / Dn);
    int d = (int)(idx % Dn);
    int s = ei[e], tt = ei[En + e], r = et[e];
    float nrm = 1.0f / counts[(long)r * Nn + tt];
    atomicAdd(&agg[(long)tt * Dn + d], h_all[((long)r * Nn + s) * Dn + d] * nrm);
}

__global__ void relu_residual_k(const float* __restrict__ agg, const float* __restrict__ xin,
                                float* __restrict__ y, long n)
{
    long i = (long)blockIdx.x * 256 + threadIdx.x;
    if (i < n) y[i] = fmaxf(agg[i], 0.f) + xin[i];
}

// partial column sums over a 128-row chunk; 3 columns per thread, coalesced rows
__global__ __launch_bounds__(256)
void col_partial_k(const float* __restrict__ y, float* __restrict__ s1, float* __restrict__ s2)
{
    const int row0 = blockIdx.x * 128;
    float a[3] = {0.f, 0.f, 0.f}, q[3] = {0.f, 0.f, 0.f};
    for (int r = 0; r < 128; ++r) {
        const float* yr = y + (long)(row0 + r) * Dn;
#pragma unroll
        for (int k = 0; k < 3; ++k) {
            float v = yr[threadIdx.x + k * 256];
            a[k] += v; q[k] += v * v;
        }
    }
#pragma unroll
    for (int k = 0; k < 3; ++k) {
        int c = threadIdx.x + k * 256;
        atomicAdd(&s1[c], a[k]);
        atomicAdd(&s2[c], q[k]);
    }
}

__global__ void col_finalize_k(const float* __restrict__ s1, const float* __restrict__ s2,
                               float* __restrict__ mean, float* __restrict__ inv)
{
    int c = blockIdx.x * 256 + threadIdx.x;
    if (c < Dn) {
        float m = s1[c] * (1.0f / Nn);
        float var = s2[c] * (1.0f / Nn) - m * m;
        mean[c] = m;
        inv[c] = rsqrtf(var + 1e-5f);
    }
}

__global__ void bn_apply_k(const float* __restrict__ y, const float* __restrict__ mean,
                           const float* __restrict__ inv, const float* __restrict__ gamma,
                           const float* __restrict__ beta, float* __restrict__ xf,
                           __bf16* __restrict__ xb, long n)
{
    long i = (long)blockIdx.x * 256 + threadIdx.x;
    if (i < n) {
        int c = (int)(i % Dn);
        float v = gamma[c] * (y[i] - mean[c]) * inv[c] + beta[c];
        xf[i] = v;
        xb[i] = (__bf16)v;
    }
}

__global__ __launch_bounds__(256)
void layernorm_rows_k(const float* __restrict__ h, const float* __restrict__ gamma,
                      const float* __restrict__ beta, float* __restrict__ out)
{
    __shared__ float s1[256], s2[256];
    const int row = blockIdx.x;
    const float* hr = h + (long)row * HIDn;
    float a = 0.f, q = 0.f;
    for (int k = threadIdx.x; k < HIDn; k += 256) {
        float v = hr[k]; a += v; q += v * v;
    }
    s1[threadIdx.x] = a; s2[threadIdx.x] = q;
    __syncthreads();
    for (int off = 128; off > 0; off >>= 1) {
        if (threadIdx.x < off) {
            s1[threadIdx.x] += s1[threadIdx.x + off];
            s2[threadIdx.x] += s2[threadIdx.x + off];
        }
        __syncthreads();
    }
    const float m = s1[0] * (1.0f / HIDn);
    const float var = s2[0] * (1.0f / HIDn) - m * m;
    const float inv = rsqrtf(var + 1e-5f);
    for (int k = threadIdx.x; k < HIDn; k += 256)
        out[(long)row * HIDn + k] = gamma[k] * (hr[k] - m) * inv + beta[k];
}

// ---------------------------------------------------------------------------
// Host-side orchestration
// ---------------------------------------------------------------------------
static inline size_t alignUp256(size_t x) { return (x + 255) & ~(size_t)255; }

extern "C" void kernel_launch(void* const* d_in, const int* in_sizes, int n_in,
                              void* d_out, int out_size, void* d_ws, size_t ws_size,
                              hipStream_t stream)
{
    const float* h_text    = (const float*)d_in[0];
    const int*   edge_idx  = (const int*)  d_in[1];
    const int*   edge_type = (const int*)  d_in[2];
    const float* w_in      = (const float*)d_in[3];
    const float* b_in      = (const float*)d_in[4];
    const float* basis     = (const float*)d_in[5];
    const float* comp      = (const float*)d_in[6];
    const float* root      = (const float*)d_in[7];
    const float* conv_bias = (const float*)d_in[8];
    const float* bn_gamma  = (const float*)d_in[9];
    const float* bn_beta   = (const float*)d_in[10];
    const float* w_out     = (const float*)d_in[11];
    const float* b_out     = (const float*)d_in[12];
    const float* ln_gamma  = (const float*)d_in[13];
    const float* ln_beta   = (const float*)d_in[14];
    float* out = (float*)d_out;

    // workspace layout
    char* w = (char*)d_ws;
    size_t off = 0;
    auto alloc = [&](size_t bytes) { void* p = w + off; off = alignUp256(off + bytes); return p; };

    __bf16* htxt_bf = (__bf16*)alloc((size_t)Nn * HIDn * 2);
    __bf16* winT    = (__bf16*)alloc((size_t)Dn * HIDn * 2);     // [D x HID]
    __bf16* woutT   = (__bf16*)alloc((size_t)HIDn * Dn * 2);     // [HID x D]
    __bf16* rootT   = (__bf16*)alloc((size_t)Dn * Dn * 2);       // [D x D]
    __bf16* WrelT   = (__bf16*)alloc((size_t)Rn * Dn * Dn * 2);  // [R x D x D]
    __bf16* xb      = (__bf16*)alloc((size_t)Nn * Dn * 2);
    float*  xf      = (float*) alloc((size_t)Nn * Dn * 4);
    float*  agg     = (float*) alloc((size_t)Nn * Dn * 4);
    float*  yb      = (float*) alloc((size_t)Nn * Dn * 4);
    float*  counts  = (float*) alloc((size_t)Rn * Nn * 4);
    float*  stats   = (float*) alloc((size_t)4 * Dn * 4);        // s1,s2,mean,inv
    float*  hfinal  = (float*) alloc((size_t)Nn * HIDn * 4);
    float*  h_all   = (float*) alloc((size_t)Rn * Nn * Dn * 4);
    (void)ws_size; (void)n_in; (void)in_sizes; (void)out_size;

    float* s1 = stats, *s2 = stats + Dn, *mean = stats + 2 * Dn, *inv = stats + 3 * Dn;

    const dim3 blk256(256);
    const dim3 gblk(256);
    auto nb = [](long n) { return (unsigned)((n + 255) / 256); };

    // 1) convert inputs
    f32_to_bf16_k<<<nb((long)Nn * HIDn), blk256, 0, stream>>>(h_text, htxt_bf, (long)Nn * HIDn);
    transpose_bf16_k<<<nb((long)HIDn * Dn), blk256, 0, stream>>>(w_in, winT, HIDn, Dn);
    transpose_bf16_k<<<nb((long)Dn * HIDn), blk256, 0, stream>>>(w_out, woutT, Dn, HIDn);

    // 2) x = relu(h_text @ w_in + b_in)
    {
        dim3 g(Dn / 128, Nn / 128, 1);
        wmma_gemm_bf16<<<g, gblk, 0, stream>>>(htxt_bf, winT, xf, b_in,
                                               Nn, Dn, HIDn, 0, 0, 1);
    }
    f32_to_bf16_k<<<nb((long)Nn * Dn), blk256, 0, stream>>>(xf, xb, (long)Nn * Dn);

    // 3) RGCN layers
    for (int l = 0; l < Ln; ++l) {
        build_rel_weights_k<<<nb((long)Rn * Dn * Dn), blk256, 0, stream>>>(
            basis + (size_t)l * Bb * Dn * Dn, comp + (size_t)l * Rn * Bb, WrelT);
        transpose_bf16_k<<<nb((long)Dn * Dn), blk256, 0, stream>>>(
            root + (size_t)l * Dn * Dn, rootT, Dn, Dn);

        // h_all[r] = x @ W_r   (batched over relations)
        {
            dim3 g(Dn / 128, Nn / 128, Rn);
            wmma_gemm_bf16<<<g, gblk, 0, stream>>>(xb, WrelT, h_all, nullptr,
                                                   Nn, Dn, Dn,
                                                   (long)Dn * Dn, (long)Nn * Dn, 0);
        }

        // per-(relation,dst) in-degree
        zero_f32_k<<<nb((long)Rn * Nn), blk256, 0, stream>>>(counts, (long)Rn * Nn);
        count_edges_k<<<nb(En), blk256, 0, stream>>>(edge_idx, edge_type, counts);

        // agg = x @ root + bias  (self-loop term, then scatter messages on top)
        {
            dim3 g(Dn / 128, Nn / 128, 1);
            wmma_gemm_bf16<<<g, gblk, 0, stream>>>(xb, rootT, agg,
                                                   conv_bias + (size_t)l * Dn,
                                                   Nn, Dn, Dn, 0, 0, 0);
        }
        scatter_msgs_k<<<nb((long)En * Dn), blk256, 0, stream>>>(
            edge_idx, edge_type, h_all, counts, agg);

        // y = relu(conv) + x ; then BatchNorm over nodes
        relu_residual_k<<<nb((long)Nn * Dn), blk256, 0, stream>>>(agg, xf, yb, (long)Nn * Dn);
        zero_f32_k<<<nb(2 * Dn), blk256, 0, stream>>>(s1, 2 * Dn);
        col_partial_k<<<Nn / 128, blk256, 0, stream>>>(yb, s1, s2);
        col_finalize_k<<<nb(Dn), blk256, 0, stream>>>(s1, s2, mean, inv);
        bn_apply_k<<<nb((long)Nn * Dn), blk256, 0, stream>>>(
            yb, mean, inv, bn_gamma + (size_t)l * Dn, bn_beta + (size_t)l * Dn,
            xf, xb, (long)Nn * Dn);
    }

    // 4) output projection + LayerNorm
    {
        dim3 g(HIDn / 128, Nn / 128, 1);
        wmma_gemm_bf16<<<g, gblk, 0, stream>>>(xb, woutT, hfinal, b_out,
                                               Nn, HIDn, Dn, 0, 0, 0);
    }
    layernorm_rows_k<<<Nn, blk256, 0, stream>>>(hfinal, ln_gamma, ln_beta, out);
}
